// PLRC_56453050138971
// MI455X (gfx1250) — matrix operand order repaired and checked by reading, not compiled, not run on hardware
//
#include <hip/hip_runtime.h>
#include <hip/hip_bf16.h>

typedef __attribute__((ext_vector_type(16))) _Float16 v16h;
typedef __attribute__((ext_vector_type(8)))  float    v8f;

#define B_   64
#define C_   256
#define HW_  56
#define P_   32
#define KEEP 16
#define CPAD 258   // 256 + 2 halves: byte stride 516 -> bank stride 129 (odd) -> conflict-free

// ---------------------------------------------------------------------------
// Kernel 1: per-batch point sampling + l2norm + WMMA similarity + argmax +
// gather.  One block per batch (grid 64, block 256 = 8 wave32).
// Outputs: out0 = pixel_feat_top [B,C,1,16], out1 = q_feats [B,C,1,16]
// ---------------------------------------------------------------------------
__global__ void plrc_points_kernel(const float* __restrict__ fxq,
                                   const float* __restrict__ fxk,
                                   const float* __restrict__ masks,
                                   float* __restrict__ out0,
                                   float* __restrict__ out1) {
    __shared__ float    s_q0[C_ * P_];     // [c][p] raw samples of fx_q
    __shared__ float    s_k0[C_ * P_];     // [c][p] raw samples of fx_k
    __shared__ int      s_off[4 * P_];     // clipped corner flat offsets
    __shared__ float    s_w[4 * P_];       // corner weights (0 if invalid)
    __shared__ float    s_rq[P_];          // 1/max(||q||,eps)
    __shared__ float    s_rk[P_];
    __shared__ _Float16 s_qh[P_ * CPAD];   // normalized q, [p][c] padded
    __shared__ _Float16 s_kh[P_ * CPAD];   // normalized k, [p][c] padded
    __shared__ float    s_sims[P_ * P_];   // sims[p][q]
    __shared__ int      s_m[P_];           // argmax over q for each p

    const int b    = blockIdx.x;
    const int t    = threadIdx.x;
    const int lane = t & 31;
    const int wave = t >> 5;

    // ---- Phase 1: bilinear corner setup for the 32 points of mask row N=0 --
    if (t < P_) {
        const int p = t;
        const float gx = masks[(b * 512 + p) * 2 + 0] * (2.0f / 55.0f) - 1.0f;
        const float gy = masks[(b * 512 + p) * 2 + 1] * (2.0f / 55.0f) - 1.0f;
        const float x = ((gx + 1.0f) * (float)HW_ - 1.0f) * 0.5f;
        const float y = ((gy + 1.0f) * (float)HW_ - 1.0f) * 0.5f;
        const float x0f = floorf(x), y0f = floorf(y);
        const float x1f = x0f + 1.0f, y1f = y0f + 1.0f;
        const int x0 = (int)x0f, y0 = (int)y0f;
        const int x1 = x0 + 1,   y1 = y0 + 1;
        const float wx1 = x1f - x, wx0 = x - x0f;   // weight toward x0 / x1
        const float wy1 = y1f - y, wy0 = y - y0f;
        const int xs[2] = {x0, x1};
        const int ys[2] = {y0, y1};
        const float fwx[2] = {wx1, wx0};
        const float fwy[2] = {wy1, wy0};
        // corner order matches reference: (x0,y0),(x0,y1),(x1,y0),(x1,y1)
#pragma unroll
        for (int ci = 0; ci < 4; ++ci) {
            const int xi = xs[ci >> 1];       // ci<2 -> x0, else x1
            const int yi = ys[ci & 1];        // even -> y0, odd -> y1
            const bool valid = (xi >= 0) && (xi <= HW_ - 1) && (yi >= 0) && (yi <= HW_ - 1);
            const int xc = min(max(xi, 0), HW_ - 1);
            const int yc = min(max(yi, 0), HW_ - 1);
            s_off[ci * P_ + p] = yc * HW_ + xc;
            s_w[ci * P_ + p]   = valid ? (fwx[ci >> 1] * fwy[ci & 1]) : 0.0f;
        }
    }
    __syncthreads();

    // ---- Phase 2: sample q0/k0, lanes sweep p (conflict-free LDS writes) ---
    for (int idx = t; idx < C_ * P_; idx += 256) {
        const int c = idx >> 5;
        const int p = idx & 31;
        const size_t base = (size_t)(b * C_ + c) * (HW_ * HW_);
        float aq = 0.0f, ak = 0.0f;
#pragma unroll
        for (int ci = 0; ci < 4; ++ci) {
            const int   o = s_off[ci * P_ + p];
            const float w = s_w[ci * P_ + p];
            aq += fxq[base + o] * w;
            ak += fxk[base + o] * w;
        }
        s_q0[idx] = aq;
        s_k0[idx] = ak;
    }
    __syncthreads();

    // ---- Phase 3: l2 norms over channels --------------------------------
    if (t < 64) {
        const int p = t & 31;
        const float* src = (t < 32) ? s_q0 : s_k0;
        float s = 0.0f;
        for (int c = 0; c < C_; ++c) {
            const float v = src[c * P_ + p];
            s += v * v;
        }
        const float r = 1.0f / fmaxf(sqrtf(s), 1e-12f);
        if (t < 32) s_rq[p] = r; else s_rk[p] = r;
    }
    __syncthreads();

    // ---- Phase 4: pack normalized features as f16, [p][c] padded ---------
    for (int idx = t; idx < C_ * P_; idx += 256) {
        const int p = idx & 31;
        const int c = idx >> 5;
        s_qh[p * CPAD + c] = (_Float16)(s_q0[c * P_ + p] * s_rq[p]);
        s_kh[p * CPAD + c] = (_Float16)(s_k0[c * P_ + p] * s_rk[p]);
    }
    __syncthreads();

    // ---- Phase 5: sims(32x32) = Kn(32x256) x Qn^T via WMMA f16 -----------
    // Waves 0..3 each own one 16x16 tile: mt = wave>>1, nt = wave&1.
    if (wave < 4) {
        const int mt = wave >> 1;
        const int nt = wave & 1;
        const int lh   = (lane < 16) ? 0 : 1;   // lane half
        const int mrow = mt * 16 + (lane & 15); // A row (pixel_top point p)
        const int ncol = nt * 16 + (lane & 15); // B col (pixel point q)
        v8f acc = {};
#pragma unroll
        for (int kb = 0; kb < 8; ++kb) {
            const int k0 = kb * 32;
            v16h a, bv;
            // A 16x32 f16 layout: lanes 0-15 hold K {0..7,16..23}, 16-31 {8..15,24..31}
            const int a0 = k0 + lh * 8;
            const int a1 = k0 + 16 + lh * 8;
#pragma unroll
            for (int j = 0; j < 8; ++j) {
                a[j]     = s_kh[mrow * CPAD + a0 + j];
                a[8 + j] = s_kh[mrow * CPAD + a1 + j];
            }
            // B 32x16 f16 layout: lanes 0-15 hold K {0..7,16..23}, 16-31 {8..15,24..31}
            const int b0 = k0 + lh * 8;
            const int b1 = k0 + 16 + lh * 8;
#pragma unroll
            for (int j = 0; j < 8; ++j) {
                bv[j]     = s_qh[ncol * CPAD + b0 + j];
                bv[8 + j] = s_qh[ncol * CPAD + b1 + j];
            }
            acc = __builtin_amdgcn_wmma_f32_16x16x32_f16(
                false, a, false, bv, (short)0, acc, false, false);
        }
        // C/D layout: VGPR v -> M = v (+8 for hi lanes), N = lane&15
#pragma unroll
        for (int v = 0; v < 8; ++v) {
            const int M = mt * 16 + v + lh * 8;
            s_sims[M * P_ + ncol] = acc[v];
        }
    }
    __syncthreads();

    // ---- Phase 6: argmax over q for each p (first-max tie rule) ----------
    if (t < P_) {
        const int p = t;
        float best = s_sims[p * P_];
        int bi = 0;
        for (int q = 1; q < P_; ++q) {
            const float v = s_sims[p * P_ + q];
            if (v > best) { best = v; bi = q; }
        }
        s_m[p] = bi;
    }
    __syncthreads();

    // ---- Phase 7: outputs (j fastest -> coalesced stores) ----------------
    for (int idx = t; idx < C_ * KEEP; idx += 256) {
        const int c = idx >> 4;
        const int j = idx & 15;
        out0[(size_t)(b * C_ + c) * KEEP + j] = s_k0[c * P_ + s_m[j]];
        out1[(size_t)(b * C_ + c) * KEEP + j] = s_q0[c * P_ + j];
    }
}

// ---------------------------------------------------------------------------
// Kernel 2: bilinear 28->56 upsample of fx_2 with NCHW -> (B*H*W, C) layout.
// Block = (b, out_row y).  LDS-stage the two input rows (stride padded to 29
// floats so per-channel column reads are bank-conflict-free); every thread
// owns one channel so the 256-float output rows store fully coalesced.
// ---------------------------------------------------------------------------
__global__ void plrc_resize_kernel(const float* __restrict__ fx2,
                                   float* __restrict__ out2) {
    __shared__ float s_rows[2 * C_ * 29];   // [r][c][xi], xi padded 28->29

    const int blk = blockIdx.x;
    const int b = blk / HW_;
    const int y = blk - b * HW_;
    const int c = threadIdx.x;

    const float sy = fmaxf(((float)y + 0.5f) * 0.5f - 0.5f, 0.0f);
    const int   yi0 = (int)floorf(sy);
    const int   yi1 = min(yi0 + 1, 27);
    const float wy  = sy - (float)yi0;

    // Stage rows yi0, yi1 of all 256 channels.
    for (int i = threadIdx.x; i < 2 * C_ * 28; i += 256) {
        const int r   = i / (C_ * 28);
        const int rem = i - r * (C_ * 28);
        const int ch  = rem / 28;
        const int xi  = rem - ch * 28;
        const int yi  = r ? yi1 : yi0;
        s_rows[r * (C_ * 29) + ch * 29 + xi] =
            fx2[((size_t)(b * C_ + ch) * 28 + yi) * 28 + xi];
    }
    __syncthreads();

    const size_t outbase = ((size_t)b * (HW_ * HW_) + (size_t)y * HW_) * C_;
    const float* r0 = &s_rows[c * 29];
    const float* r1 = &s_rows[C_ * 29 + c * 29];
#pragma unroll 4
    for (int x = 0; x < HW_; ++x) {
        const float sx = fmaxf(((float)x + 0.5f) * 0.5f - 0.5f, 0.0f);
        const int   xi0 = (int)floorf(sx);
        const int   xi1 = min(xi0 + 1, 27);
        const float wx  = sx - (float)xi0;
        const float f0 = r0[xi0] * (1.0f - wy) + r1[xi0] * wy;  // y-interp @ xi0
        const float f1 = r0[xi1] * (1.0f - wy) + r1[xi1] * wy;  // y-interp @ xi1
        out2[outbase + (size_t)x * C_ + c] = f0 * (1.0f - wx) + f1 * wx;
    }
}

// ---------------------------------------------------------------------------
extern "C" void kernel_launch(void* const* d_in, const int* in_sizes, int n_in,
                              void* d_out, int out_size, void* d_ws, size_t ws_size,
                              hipStream_t stream) {
    (void)in_sizes; (void)n_in; (void)d_ws; (void)ws_size; (void)out_size;
    const float* fxq   = (const float*)d_in[0];
    const float* fxk   = (const float*)d_in[1];
    const float* fx2   = (const float*)d_in[2];
    const float* masks = (const float*)d_in[3];
    // d_in[4] = keep_point (== 16, baked into output shapes)

    float* out  = (float*)d_out;
    float* out0 = out;                       // pixel_feat_top [64,256,1,16]
    float* out1 = out + (size_t)B_ * C_ * KEEP;       // q_feats
    float* out2 = out + (size_t)2 * B_ * C_ * KEEP;   // fx2_flat [64*3136, 256]

    plrc_points_kernel<<<B_, 256, 0, stream>>>(fxq, fxk, masks, out0, out1);
    plrc_resize_kernel<<<B_ * HW_, 256, 0, stream>>>(fx2, out2);
}
